// basket_GRU_MI_E_76295799046567
// MI455X (gfx1250) — compile-verified, matrix operand order checked by reading, not run on hardware
//
#include <hip/hip_runtime.h>

// ---------------- problem dims ----------------
constexpr int B_DIM = 64;
constexpr int T_DIM = 128;
constexpr int D_DIM = 4096;
constexpr int NH    = 1024;
constexpr int DD    = 512;
constexpr int MTOT  = B_DIM * T_DIM;   // 8192 rows (b*T + t)
constexpr int CTOT  = D_DIM + DD;      // 4608 concat width

// ---------------- vector types ----------------
typedef __attribute__((ext_vector_type(16))) __bf16 bf16x16;
typedef __attribute__((ext_vector_type(8)))  __bf16 bf16x8;
typedef __attribute__((ext_vector_type(4)))  __bf16 bf16x4;
typedef __attribute__((ext_vector_type(8)))  float  f32x8;
typedef __attribute__((ext_vector_type(4)))  float  f32x4;
typedef __attribute__((ext_vector_type(4)))  unsigned int u32x4;
typedef __attribute__((ext_vector_type(8)))  int    i32x8;
typedef __attribute__((ext_vector_type(4)))  int    i32x4;

#define WMMA_BF16(a, b, c) \
  __builtin_amdgcn_wmma_f32_16x16x32_bf16(false, (a), false, (b), (short)0, (c), false, false)

// 16x32 bf16 fragment load (A, or B from a [n][k] transposed array).
// ISA 7.12.2: lane L -> row M=L&15; lane-half selects K base (0 / 8), VGPR
// halves hold K+0..7 and K+16..23.  Two 16B loads per fragment.
__device__ inline bf16x16 load_frag(const __bf16* p) {
  bf16x8 lo = *(const bf16x8*)(p);
  bf16x8 hi = *(const bf16x8*)(p + 16);
  bf16x16 r;
#pragma unroll
  for (int i = 0; i < 8; ++i) { r[i] = lo[i]; r[i + 8] = hi[i]; }
  return r;
}

__device__ inline f32x8 zero_f32x8() {
  f32x8 z;
#pragma unroll
  for (int i = 0; i < 8; ++i) z[i] = 0.0f;
  return z;
}

__device__ inline float sigmoidf(float v) { return 1.0f / (1.0f + __expf(-v)); }

// ---------------------------------------------------------------------------
// Tensor Data Mover: 2D bf16 tile (tile_d0 along K, tile_d1 rows) from global
// to LDS, with LDS row padding 64B->80B (pad_interval=16 DW, pad_amount=4 DW).
// Descriptor packing per CDNA5 ISA.
// ---------------------------------------------------------------------------
__device__ inline void tdm_issue(u32x4 g0, i32x8 g1) {
#if defined(__clang_major__) && (__clang_major__ >= 23)
  __builtin_amdgcn_tensor_load_to_lds(g0, g1, (i32x4)0, (i32x4)0, (i32x8)0, 0);
#else
  __builtin_amdgcn_tensor_load_to_lds(g0, g1, (i32x4)0, (i32x4)0, 0);
#endif
}

__device__ inline void tdm_load_tile_2d(const __bf16* gptr, unsigned lds_byte_off,
                                        unsigned tensor_d0, unsigned tensor_d1,
                                        unsigned long long stride0,
                                        unsigned tile_d0, unsigned tile_d1) {
  unsigned long long ga = (unsigned long long)(size_t)gptr;
  u32x4 g0;
  g0[0] = 1u;                                            // count=1 user D#
  g0[1] = lds_byte_off;                                  // lds_addr
  g0[2] = (unsigned)(ga & 0xFFFFFFFFu);                  // global_addr lo
  g0[3] = (unsigned)((ga >> 32) & 0x01FFFFFFu) | (2u << 30);  // hi + type=2
  i32x8 g1;
  const unsigned data_size    = 1;  // 2-byte elements
  const unsigned pad_enable   = 1;
  const unsigned pad_interval = 3;  // 16 DWORDs = one 32-elem bf16 row
  const unsigned pad_amount   = 3;  // 4 DWORDs  -> 80B LDS row stride
  g1[0] = (int)((data_size << 16) | (pad_enable << 20) |
                (pad_interval << 22) | (pad_amount << 25));
  g1[1] = (int)((tensor_d0 & 0xFFFFu) << 16);
  g1[2] = (int)(((tensor_d0 >> 16) & 0xFFFFu) | ((tensor_d1 & 0xFFFFu) << 16));
  g1[3] = (int)(((tensor_d1 >> 16) & 0xFFFFu) | ((tile_d0 & 0xFFFFu) << 16));
  g1[4] = (int)(tile_d1 & 0xFFFFu);
  g1[5] = (int)(stride0 & 0xFFFFFFFFull);
  g1[6] = (int)((stride0 >> 32) & 0xFFFFull);
  g1[7] = 0;
  tdm_issue(g0, g1);
}

// ---------------------------------------------------------------------------
// Pack kernels: build bf16 operands once (float4 reads, 8B bf16 writes).
// ---------------------------------------------------------------------------
__global__ __launch_bounds__(256) void pack_cat_kernel(
    const float* __restrict__ x, const float* __restrict__ fi,
    __bf16* __restrict__ catB) {
  size_t idx = (size_t)blockIdx.x * blockDim.x + threadIdx.x;  // one per 4 elems
  constexpr size_t QW = (size_t)CTOT / 4;
  if (idx >= (size_t)MTOT * QW) return;
  int m  = (int)(idx / QW);
  int c4 = (int)(idx % QW) * 4;
  f32x4 v;
  if (c4 < D_DIM) v = *(const f32x4*)&x[(size_t)m * D_DIM + c4];
  else            v = *(const f32x4*)&fi[((size_t)m * 2 + 1) * D_DIM + (c4 - D_DIM)];
  bf16x4 o;
#pragma unroll
  for (int i = 0; i < 4; ++i) o[i] = (__bf16)v[i];
  *(bf16x4*)&catB[(size_t)m * CTOT + c4] = o;
}

__global__ __launch_bounds__(256) void pack_dot_kernel(
    const float* __restrict__ fo, __bf16* __restrict__ dotB) {
  size_t idx = (size_t)blockIdx.x * blockDim.x + threadIdx.x;
  constexpr size_t QW = (size_t)DD / 4;
  if (idx >= (size_t)MTOT * QW) return;
  int m  = (int)(idx / QW);
  int c4 = (int)(idx % QW) * 4;
  f32x4 v = *(const f32x4*)&fo[((size_t)m * 2 + 1) * D_DIM + c4];
  bf16x4 o;
#pragma unroll
  for (int i = 0; i < 4; ++i) o[i] = (__bf16)v[i];
  *(bf16x4*)&dotB[(size_t)m * DD + c4] = o;
}

// (K,N) f32 row-major -> (N,K) bf16 row-major (coalesced read along N).
__global__ __launch_bounds__(256) void convert_wt_kernel(
    const float* __restrict__ in, __bf16* __restrict__ out, int K, int N) {
  size_t idx = (size_t)blockIdx.x * blockDim.x + threadIdx.x;
  if (idx < (size_t)K * N) {
    int k = (int)(idx / N);
    int n = (int)(idx % N);
    out[(size_t)n * K + k] = (__bf16)in[idx];
  }
}

// ---------------------------------------------------------------------------
// Kernel A: P = catB @ W + bias  (bf16 in, f32 accum, bf16 out)
// 128x128 tile, 8 waves (4M x 2N).  A and B tiles streamed into LDS by the
// Tensor Data Mover, double-buffered, synchronized via TENSORcnt + barrier.
// ---------------------------------------------------------------------------
__global__ __launch_bounds__(256) void pre_gemm_kernel(
    const __bf16* __restrict__ catB, const __bf16* __restrict__ Wt,
    const float* __restrict__ bias, __bf16* __restrict__ P) {
  constexpr int KB  = 32;
  constexpr int LDA = KB + 8;  // 80B rows (matches TDM pad), 16B aligned
  __shared__ __bf16 Abuf[2][128 * LDA];
  __shared__ __bf16 Bbuf[2][128 * LDA];

  const int m0 = blockIdx.x * 128;
  const int n0 = blockIdx.y * 128;
  const int tid  = threadIdx.x;
  const int lane = tid & 31;
  const int wave = tid >> 5;
  const int wm = wave >> 1;          // 0..3 -> 32-row block
  const int wn = wave & 1;           // 0..1 -> 64-col block
  const int fragM = lane & 15;
  const int fragK = (lane >> 4) * 8;
  const int accRow = (lane >> 4) * 8;
  const int accCol = lane & 15;

  f32x8 acc[2][4];
#pragma unroll
  for (int i = 0; i < 2; ++i)
#pragma unroll
    for (int j = 0; j < 4; ++j) acc[i][j] = zero_f32x8();

  const int nk = CTOT / KB;  // 144
  // TDM stage of K-chunk ks into buffer ks&1 (wave 0 issues; EXEC ignored)
  auto stage = [&](int ks) {
    int buf = ks & 1;
    tdm_load_tile_2d(catB + (size_t)m0 * CTOT + ks * KB,
                     (unsigned)(size_t)&Abuf[buf][0],
                     CTOT, MTOT, CTOT, KB, 128);
    tdm_load_tile_2d(Wt + (size_t)n0 * CTOT + ks * KB,
                     (unsigned)(size_t)&Bbuf[buf][0],
                     CTOT, NH, CTOT, KB, 128);
  };
  if (wave == 0) stage(0);

  for (int ks = 0; ks < nk; ++ks) {
    if (wave == 0) __builtin_amdgcn_s_wait_tensorcnt((short)0);
    __syncthreads();                       // publish tile ks; prior reads done
    if (wave == 0 && ks + 1 < nk) stage(ks + 1);

    const __bf16* Ab = &Abuf[ks & 1][0];
    const __bf16* Bb = &Bbuf[ks & 1][0];
    bf16x16 af[2], bf[4];
#pragma unroll
    for (int i = 0; i < 2; ++i)
      af[i] = load_frag(&Ab[(wm * 32 + i * 16 + fragM) * LDA + fragK]);
#pragma unroll
    for (int j = 0; j < 4; ++j)
      bf[j] = load_frag(&Bb[(wn * 64 + j * 16 + fragM) * LDA + fragK]);
#pragma unroll
    for (int i = 0; i < 2; ++i)
#pragma unroll
      for (int j = 0; j < 4; ++j) acc[i][j] = WMMA_BF16(af[i], bf[j], acc[i][j]);
  }

#pragma unroll
  for (int i = 0; i < 2; ++i)
#pragma unroll
    for (int j = 0; j < 4; ++j)
#pragma unroll
      for (int v = 0; v < 8; ++v) {
        int r = m0 + wm * 32 + i * 16 + accRow + v;
        int c = n0 + wn * 64 + j * 16 + accCol;
        P[(size_t)r * NH + c] = (__bf16)(acc[i][j][v] + bias[c]);
      }
}

// ---------------------------------------------------------------------------
// Kernel B: gated recurrence.  4 blocks x 16 batch rows, 16 waves x 64 cols.
// h: f32 in registers (accumulator layout) + bf16 in one LDS buffer cycling
// h -> h*s_g -> h_new.  Weight fragments straight from L2-resident bf16ᵀ.
// ---------------------------------------------------------------------------
__global__ __launch_bounds__(512) void recurrence_kernel(
    const __bf16* __restrict__ pf, const __bf16* __restrict__ ps,
    const __bf16* __restrict__ pi, const __bf16* __restrict__ Wfa_t,
    const __bf16* __restrict__ Wsa_t, const __bf16* __restrict__ Wia_t,
    __bf16* __restrict__ Hb) {
  constexpr int ROWS = 16;
  constexpr int LH = NH + 8;  // 2064B rows, 16B aligned, conflict pad
  __shared__ __bf16 hb[ROWS * LH];

  const int b0 = blockIdx.x * ROWS;
  const int tid  = threadIdx.x;
  const int lane = tid & 31;
  const int wave = tid >> 5;
  const int n0 = wave * 64;
  const int fragM = lane & 15;
  const int fragK = (lane >> 4) * 8;
  const int accRow = (lane >> 4) * 8;
  const int accCol = lane & 15;

  for (int i = tid; i < ROWS * LH; i += blockDim.x) hb[i] = (__bf16)0.0f;
  float hreg[4][8];
#pragma unroll
  for (int j = 0; j < 4; ++j)
#pragma unroll
    for (int v = 0; v < 8; ++v) hreg[j][v] = 0.0f;
  __syncthreads();

  for (int t = 0; t < T_DIM; ++t) {
    f32x8 accF[4], accS[4];
#pragma unroll
    for (int j = 0; j < 4; ++j) { accF[j] = zero_f32x8(); accS[j] = zero_f32x8(); }
    for (int k0 = 0; k0 < NH; k0 += 32) {
      bf16x16 a = load_frag(&hb[fragM * LH + k0 + fragK]);
#pragma unroll
      for (int j = 0; j < 4; ++j) {
        int n = n0 + j * 16 + fragM;
        bf16x16 bF = load_frag(&Wfa_t[(size_t)n * NH + k0 + fragK]);
        bf16x16 bS = load_frag(&Wsa_t[(size_t)n * NH + k0 + fragK]);
        accF[j] = WMMA_BF16(a, bF, accF[j]);
        accS[j] = WMMA_BF16(a, bS, accS[j]);
      }
    }
    __syncthreads();  // done reading hb (= h_t)

    float fgate[4][8];
#pragma unroll
    for (int j = 0; j < 4; ++j)
#pragma unroll
      for (int v = 0; v < 8; ++v) {
        int r = accRow + v;
        int c = n0 + j * 16 + accCol;
        size_t p = ((size_t)(b0 + r) * T_DIM + t) * NH + c;
        float f = sigmoidf((float)pf[p] + accF[j][v]);
        float s = sigmoidf((float)ps[p] + accS[j][v]);
        fgate[j][v] = f;
        hb[r * LH + c] = (__bf16)(hreg[j][v] * s);  // hb := h * s_g
      }
    __syncthreads();

    f32x8 accI[4];
#pragma unroll
    for (int j = 0; j < 4; ++j) accI[j] = zero_f32x8();
    for (int k0 = 0; k0 < NH; k0 += 32) {
      bf16x16 a = load_frag(&hb[fragM * LH + k0 + fragK]);
#pragma unroll
      for (int j = 0; j < 4; ++j) {
        int n = n0 + j * 16 + fragM;
        bf16x16 bI = load_frag(&Wia_t[(size_t)n * NH + k0 + fragK]);
        accI[j] = WMMA_BF16(a, bI, accI[j]);
      }
    }
    __syncthreads();  // done reading hb (= h*s_g)

#pragma unroll
    for (int j = 0; j < 4; ++j)
#pragma unroll
      for (int v = 0; v < 8; ++v) {
        int r = accRow + v;
        int c = n0 + j * 16 + accCol;
        size_t p = ((size_t)(b0 + r) * T_DIM + t) * NH + c;
        float cc = tanhf((float)pi[p] + accI[j][v]);
        float f = fgate[j][v];
        float hn = hreg[j][v] * (1.0f - f) + f * cc;
        hreg[j][v] = hn;
        hb[r * LH + c] = (__bf16)hn;   // hb := h_{t+1}
        Hb[p] = (__bf16)hn;
      }
    __syncthreads();
  }
}

// ---------------------------------------------------------------------------
// Kernel C: out = sigmoid( (H @ W_ba) * (1 + dot @ E) + e_o )
// All operands bf16 with K contiguous -> zero LDS, fragments via b128 global
// loads (Hb/dotB stream, weightsᵀ L2-resident).
// ---------------------------------------------------------------------------
__global__ __launch_bounds__(256) void out_gemm_kernel(
    const __bf16* __restrict__ Hb, const __bf16* __restrict__ dotB,
    const __bf16* __restrict__ Wba_t, const __bf16* __restrict__ Et,
    const float* __restrict__ e_o, float* __restrict__ out) {
  const int m0 = blockIdx.x * 128;
  const int n0 = blockIdx.y * 128;
  const int tid  = threadIdx.x;
  const int lane = tid & 31;
  const int wave = tid >> 5;
  const int wm = wave >> 1;
  const int wn = wave & 1;
  const int fragM = lane & 15;
  const int fragK = (lane >> 4) * 8;
  const int accRow = (lane >> 4) * 8;
  const int accCol = lane & 15;

  f32x8 acc1[2][4], acc2[2][4];
#pragma unroll
  for (int i = 0; i < 2; ++i)
#pragma unroll
    for (int j = 0; j < 4; ++j) { acc1[i][j] = zero_f32x8(); acc2[i][j] = zero_f32x8(); }

  // pass 1: H @ W_ba  (K = NH)
  for (int k0 = 0; k0 < NH; k0 += 32) {
    bf16x16 af[2], bf[4];
#pragma unroll
    for (int i = 0; i < 2; ++i) {
      int m = m0 + wm * 32 + i * 16 + fragM;
      af[i] = load_frag(&Hb[(size_t)m * NH + k0 + fragK]);
    }
#pragma unroll
    for (int j = 0; j < 4; ++j) {
      int n = n0 + wn * 64 + j * 16 + fragM;
      bf[j] = load_frag(&Wba_t[(size_t)n * NH + k0 + fragK]);
    }
#pragma unroll
    for (int i = 0; i < 2; ++i)
#pragma unroll
      for (int j = 0; j < 4; ++j) acc1[i][j] = WMMA_BF16(af[i], bf[j], acc1[i][j]);
  }

  // pass 2: dot @ E  (K = DD)
  for (int k0 = 0; k0 < DD; k0 += 32) {
    bf16x16 af[2], bf[4];
#pragma unroll
    for (int i = 0; i < 2; ++i) {
      int m = m0 + wm * 32 + i * 16 + fragM;
      af[i] = load_frag(&dotB[(size_t)m * DD + k0 + fragK]);
    }
#pragma unroll
    for (int j = 0; j < 4; ++j) {
      int n = n0 + wn * 64 + j * 16 + fragM;
      bf[j] = load_frag(&Et[(size_t)n * DD + k0 + fragK]);
    }
#pragma unroll
    for (int i = 0; i < 2; ++i)
#pragma unroll
      for (int j = 0; j < 4; ++j) acc2[i][j] = WMMA_BF16(af[i], bf[j], acc2[i][j]);
  }

#pragma unroll
  for (int i = 0; i < 2; ++i)
#pragma unroll
    for (int j = 0; j < 4; ++j)
#pragma unroll
      for (int v = 0; v < 8; ++v) {
        int r = m0 + wm * 32 + i * 16 + accRow + v;
        int c = n0 + wn * 64 + j * 16 + accCol;
        float val = acc1[i][j][v] * (1.0f + acc2[i][j][v]) + e_o[c];
        out[(size_t)r * D_DIM + c] = sigmoidf(val);
      }
}

// ---------------------------------------------------------------------------
extern "C" void kernel_launch(void* const* d_in, const int* in_sizes, int n_in,
                              void* d_out, int out_size, void* d_ws, size_t ws_size,
                              hipStream_t stream) {
  const float* x    = (const float*)d_in[0];
  const float* fo   = (const float*)d_in[1];
  const float* fi   = (const float*)d_in[2];
  const float* E    = (const float*)d_in[3];
  const float* W_fb = (const float*)d_in[4];
  const float* W_fa = (const float*)d_in[5];
  const float* e_f  = (const float*)d_in[6];
  const float* W_ib = (const float*)d_in[7];
  const float* W_ia = (const float*)d_in[8];
  const float* e_i  = (const float*)d_in[9];
  const float* W_sb = (const float*)d_in[10];
  const float* W_sa = (const float*)d_in[11];
  const float* e_s  = (const float*)d_in[12];
  const float* W_ba = (const float*)d_in[13];
  const float* e_o  = (const float*)d_in[14];
  float* out = (float*)d_out;

  // workspace carve-up (all bf16 intermediates; ~197 MB total)
  char* ws = (char*)d_ws;
  auto take = [&](size_t elems) { __bf16* p = (__bf16*)ws; ws += elems * 2; return p; };
  __bf16* catB = take((size_t)MTOT * CTOT);
  __bf16* dotB = take((size_t)MTOT * DD);
  __bf16* pfB  = take((size_t)MTOT * NH);
  __bf16* psB  = take((size_t)MTOT * NH);
  __bf16* piB  = take((size_t)MTOT * NH);
  __bf16* Hb   = take((size_t)MTOT * NH);
  __bf16* Wfbt = take((size_t)NH * CTOT);
  __bf16* Wsbt = take((size_t)NH * CTOT);
  __bf16* Wibt = take((size_t)NH * CTOT);
  __bf16* Wfat = take((size_t)NH * NH);
  __bf16* Wsat = take((size_t)NH * NH);
  __bf16* Wiat = take((size_t)NH * NH);
  __bf16* Wbat = take((size_t)D_DIM * NH);
  __bf16* Et   = take((size_t)D_DIM * DD);

  // pack activations to bf16
  {
    size_t n1 = (size_t)MTOT * (CTOT / 4);
    pack_cat_kernel<<<(unsigned)((n1 + 255) / 256), 256, 0, stream>>>(x, fi, catB);
    size_t n2 = (size_t)MTOT * (DD / 4);
    pack_dot_kernel<<<(unsigned)((n2 + 255) / 256), 256, 0, stream>>>(fo, dotB);
  }
  // transpose+convert weights to bf16
  auto cv = [&](const float* in, __bf16* o, int K, int N) {
    size_t tot = (size_t)K * N;
    convert_wt_kernel<<<(unsigned)((tot + 255) / 256), 256, 0, stream>>>(in, o, K, N);
  };
  cv(W_fb, Wfbt, CTOT, NH);
  cv(W_sb, Wsbt, CTOT, NH);
  cv(W_ib, Wibt, CTOT, NH);
  cv(W_fa, Wfat, NH, NH);
  cv(W_sa, Wsat, NH, NH);
  cv(W_ia, Wiat, NH, NH);
  cv(W_ba, Wbat, NH, D_DIM);
  cv(E,    Et,   DD, D_DIM);

  dim3 gA(MTOT / 128, NH / 128, 1);
  pre_gemm_kernel<<<gA, 256, 0, stream>>>(catB, Wfbt, e_f, pfB);
  pre_gemm_kernel<<<gA, 256, 0, stream>>>(catB, Wsbt, e_s, psB);
  pre_gemm_kernel<<<gA, 256, 0, stream>>>(catB, Wibt, e_i, piB);

  recurrence_kernel<<<B_DIM / 16, 512, 0, stream>>>(pfB, psB, piB, Wfat, Wsat,
                                                    Wiat, Hb);

  dim3 gC(MTOT / 128, D_DIM / 128, 1);
  out_gemm_kernel<<<gC, 256, 0, stream>>>(Hb, dotB, Wbat, Et, e_o, out);
}
